// GATJK_52441550684201
// MI455X (gfx1250) — matrix operand order repaired and checked by reading, not compiled, not run on hardware
//
#include <hip/hip_runtime.h>
#include <hip/hip_bf16.h>

#define NN   40000
#define EE   320000
#define FIN  128
#define HEADS 4
#define CH   64
#define HC   256
#define OUTC 64
#define NH   (NN * HEADS)

typedef __attribute__((ext_vector_type(16))) __bf16 v16bf;
typedef __attribute__((ext_vector_type(8)))  float  v8f;
typedef unsigned int v4u __attribute__((ext_vector_type(4)));
typedef int v4i __attribute__((ext_vector_type(4)));
typedef int v8i __attribute__((ext_vector_type(8)));

union FragBF { unsigned int u[8]; v16bf v; };

// Tensor Data Mover available? (device pass on gfx1250 toolchains; host pass -> 0)
#if __has_builtin(__builtin_amdgcn_tensor_load_to_lds)
#define USE_TDM 1
#else
#define USE_TDM 0
#endif

__device__ __forceinline__ unsigned short f2bf(float f) {
    unsigned u = __float_as_uint(f);
    unsigned r = u + 0x7FFFu + ((u >> 16) & 1u);   // round-to-nearest-even
    return (unsigned short)(r >> 16);
}
__device__ __forceinline__ float bf2f(unsigned short b) {
    return __uint_as_float(((unsigned)b) << 16);
}
// order-preserving float <-> uint key for atomicMax segment-max
__device__ __forceinline__ unsigned fkey(float f) {
    unsigned u = __float_as_uint(f);
    return (u & 0x80000000u) ? ~u : (u | 0x80000000u);
}
__device__ __forceinline__ float fdec(unsigned k) {
    unsigned u = (k & 0x80000000u) ? (k & 0x7FFFFFFFu) : ~k;
    return __uint_as_float(u);
}

// --------------------------------------------------------------- fp32 -> bf16
__global__ void k_cvt_bf16(const float* __restrict__ src,
                           unsigned short* __restrict__ dst, int n) {
    int i = blockIdx.x * blockDim.x + threadIdx.x;
    if (i < n) dst[i] = f2bf(src[i]);
}

// ------------------------------------------------- pack W into WMMA B-fragment
// layout: Bp[(kt*(N/16)+jt)*32*8 + lane*8 + v] = bf16(W[k][col]) | bf16(W[k+1][col])<<16
// with col = jt*16 + (lane&15), half = lane>>4, k = kt*32 + 2v + half*16
__global__ void k_pack_b(const float* __restrict__ W, unsigned* __restrict__ Bp,
                         int K, int N) {
    int idx = blockIdx.x * blockDim.x + threadIdx.x;
    if (idx >= (K * N) / 2) return;
    int v = idx & 7;
    int lane = (idx >> 3) & 31;
    int tile = idx >> 8;
    int jt = tile % (N >> 4);
    int kt = tile / (N >> 4);
    int col = jt * 16 + (lane & 15);
    int half = lane >> 4;
    int k = kt * 32 + 2 * v + half * 16;
    unsigned lo = f2bf(W[(size_t)k * N + col]);
    unsigned hi = f2bf(W[(size_t)(k + 1) * N + col]);
    Bp[idx] = lo | (hi << 16);
}

#if USE_TDM
// Issue one 128 x 32 bf16 tile load (global -> LDS) through the TDM.
__device__ __forceinline__ void tdm_issue(const unsigned short* Ab, int M, int rowBase,
                                          int kt, int kdim, unsigned short* dstLds) {
    unsigned long long ga =
        (unsigned long long)(uintptr_t)(Ab + (size_t)rowBase * kdim + kt * 32);
    unsigned ldsOff = (unsigned)(uintptr_t)dstLds;          // low 32 bits = LDS offset
    unsigned rows = (unsigned)(M - rowBase);
    // D# group0: count=1 | lds_addr | global_addr | type=2
    v4u g0 = { 1u, ldsOff, (unsigned)ga,
               (unsigned)((ga >> 32) & 0x01FFFFFFull) | 0x80000000u };
    // D# group1: data_size=2B; tensor_dim0=kdim, tensor_dim1=rows remaining;
    //            tile 32 x 128; tensor_dim0_stride=kdim
    v8i g1 = { (int)(1u << 16),
               (int)(((unsigned)kdim & 0xFFFFu) << 16),
               (int)(((unsigned)kdim >> 16) | ((rows & 0xFFFFu) << 16)),
               (int)((rows >> 16) | (32u << 16)),
               (int)128u,
               (int)(unsigned)kdim,
               0, 0 };
    v4i z4 = { 0, 0, 0, 0 };
#if defined(__clang_major__) && (__clang_major__ >= 23)
    v8i z8 = { 0, 0, 0, 0, 0, 0, 0, 0 };
    __builtin_amdgcn_tensor_load_to_lds(g0, g1, z4, z4, z8, 0);
#else
    __builtin_amdgcn_tensor_load_to_lds(g0, g1, z4, z4, 0);
#endif
}
#endif

// one K-step of MMA: A fragment = 2x ds_load_b128; B fragments = 2x global b128 each
__device__ __forceinline__ void tile_mma(const unsigned short* __restrict__ buf,
                                         const unsigned* __restrict__ Bp,
                                         int kt, int ntiles16, int colTile0,
                                         int wr, int li, int half, int lane,
                                         v8f acc[4]) {
    FragBF fa;
    *(uint4*)&fa.u[0] = *(const uint4*)&buf[(wr + li) * 32 + half * 16];
    *(uint4*)&fa.u[4] = *(const uint4*)&buf[(wr + li) * 32 + half * 16 + 8];
    #pragma unroll
    for (int j = 0; j < 4; ++j) {
        FragBF fb;
        const uint4* pb = (const uint4*)(Bp +
            ((size_t)(kt * ntiles16 + colTile0 + j) * 32 + lane) * 8);
        *(uint4*)&fb.u[0] = pb[0];
        *(uint4*)&fb.u[4] = pb[1];
        acc[j] = __builtin_amdgcn_wmma_f32_16x16x32_bf16(
            false, fa.v, false, fb.v, (short)0, acc[j], false, false);
    }
}

// ---------------------------------------------------------------- WMMA GEMM
// C[M x NFULL] (fp32) = Ab[M x KDIM] (bf16) @ Bp (pre-packed bf16 fragments)
template<int KDIM, int NFULL>
__global__ __launch_bounds__(256) void k_gemm(
        const unsigned short* __restrict__ Ab, const unsigned* __restrict__ Bp,
        const float* __restrict__ bias, float* __restrict__ C, int M)
{
    __shared__ __align__(16) unsigned short sA0[128 * 32];  // 8KB double buffer
    __shared__ __align__(16) unsigned short sA1[128 * 32];

    const int t = threadIdx.x;
    const int rowBase = blockIdx.x * 128;
    const int colTile0 = blockIdx.y * 4;            // 16-col tiles
    const int wave = t >> 5, lane = t & 31;
    const int half = lane >> 4, li = lane & 15;
    const int wr = wave * 16;
    const int ntiles16 = NFULL / 16;
    const int T = KDIM / 32;

    v8f acc[4];
    #pragma unroll
    for (int j = 0; j < 4; ++j)
        #pragma unroll
        for (int v = 0; v < 8; ++v) acc[j][v] = 0.0f;

#if USE_TDM
    if (t < 32) tdm_issue(Ab, M, rowBase, 0, KDIM, sA0);
    for (int kt = 0; kt < T; ++kt) {
        unsigned short* cur = (kt & 1) ? sA1 : sA0;
        unsigned short* nxt = (kt & 1) ? sA0 : sA1;
        if (t < 32) {
            if (kt + 1 < T) {
                tdm_issue(Ab, M, rowBase, kt + 1, KDIM, nxt);  // overlap next DMA
                __builtin_amdgcn_s_wait_tensorcnt(1);
            } else {
                __builtin_amdgcn_s_wait_tensorcnt(0);
            }
        }
        __syncthreads();
        tile_mma(cur, Bp, kt, ntiles16, colTile0, wr, li, half, lane, acc);
        __syncthreads();
    }
#else
    for (int kt = 0; kt < T; ++kt) {
        unsigned short* cur = (kt & 1) ? sA1 : sA0;
        #pragma unroll
        for (int i = 0; i < 2; ++i) {                // 8KB / 256 thr = 32B each
            int lin = t + i * 256;                   // 0..511
            int r = lin >> 2;                        // 0..127
            int g = lin & 3;                         // 4 groups of 8 bf16
            int gr = rowBase + r; if (gr >= M) gr = M - 1;
            *(uint4*)&cur[r * 32 + g * 8] =
                *(const uint4*)(Ab + (size_t)gr * KDIM + kt * 32 + g * 8);
        }
        __syncthreads();
        tile_mma(cur, Bp, kt, ntiles16, colTile0, wr, li, half, lane, acc);
        __syncthreads();
    }
#endif

    // epilogue: D layout M = v + half*8, N = li
    #pragma unroll
    for (int j = 0; j < 4; ++j) {
        int col = colTile0 * 16 + j * 16 + li;
        #pragma unroll
        for (int v = 0; v < 8; ++v) {
            int gr = rowBase + wr + v + half * 8;
            if (gr < M) {
                float val = acc[j][v];
                if (bias) val += bias[col];
                C[(size_t)gr * NFULL + col] = val;
            }
        }
    }
}

// ------------------------------------------------- per-node attention scores
__global__ __launch_bounds__(128) void k_attn(
        const float* __restrict__ feat, const float* __restrict__ asrc,
        const float* __restrict__ adst, float* __restrict__ esrc,
        float* __restrict__ edst)
{
    int n = blockIdx.x;
    int w = threadIdx.x >> 5, lane = threadIdx.x & 31;
    const float* f = feat + (size_t)n * HC + w * CH;
    int c = lane * 2;
    float s = f[c] * asrc[w * CH + c] + f[c + 1] * asrc[w * CH + c + 1];
    float d = f[c] * adst[w * CH + c] + f[c + 1] * adst[w * CH + c + 1];
    #pragma unroll
    for (int o = 16; o; o >>= 1) { s += __shfl_xor(s, o, 32); d += __shfl_xor(d, o, 32); }
    if (lane == 0) { esrc[n * HEADS + w] = s; edst[n * HEADS + w] = d; }
}

__global__ void k_init_attn(unsigned* __restrict__ umax, float* __restrict__ denom) {
    int i = blockIdx.x * blockDim.x + threadIdx.x;
    if (i < NH) { umax[i] = 0u; denom[i] = 0.0f; }
}

__global__ void k_edge1(const float* __restrict__ esrc, const float* __restrict__ edst,
                        const long long* __restrict__ ei, float* __restrict__ ee,
                        unsigned* __restrict__ umax)
{
    int tid = blockIdx.x * blockDim.x + threadIdx.x;
    if (tid >= EE * HEADS) return;
    int e = tid >> 2, h = tid & 3;
    int s = (int)ei[e], d = (int)ei[EE + e];
    float v = esrc[s * HEADS + h] + edst[d * HEADS + h];
    float lr = v > 0.0f ? v : 0.2f * v;
    ee[tid] = lr;
    atomicMax(&umax[d * HEADS + h], fkey(lr));
}

__global__ void k_edge2(const long long* __restrict__ ei, float* __restrict__ ee,
                        const unsigned* __restrict__ umax, float* __restrict__ denom)
{
    int tid = blockIdx.x * blockDim.x + threadIdx.x;
    if (tid >= EE * HEADS) return;
    int e = tid >> 2, h = tid & 3;
    int d = (int)ei[EE + e];
    float p = __expf(ee[tid] - fdec(umax[d * HEADS + h]));
    ee[tid] = p;
    atomicAdd(&denom[d * HEADS + h], p);
}

__global__ void k_agg_init(float* __restrict__ agg, const float* __restrict__ bias) {
    int i = blockIdx.x * blockDim.x + threadIdx.x;
    if (i < NN * HC) agg[i] = bias[i & (HC - 1)];
}

__global__ __launch_bounds__(256) void k_scatter(
        const float* __restrict__ feat, const float* __restrict__ ee,
        const float* __restrict__ denom, const long long* __restrict__ ei,
        float* __restrict__ agg)
{
    int e = blockIdx.x, t = threadIdx.x;
    int s = (int)ei[e], d = (int)ei[EE + e];
    if (e + 1 < EE) {   // prefetch next edge's source feature row (global_prefetch_b8)
        int sn = (int)ei[e + 1];
        __builtin_prefetch(feat + (size_t)sn * HC + t, 0, 1);
    }
    int h = t >> 6;
    float alpha = ee[e * HEADS + h] / (denom[d * HEADS + h] + 1e-16f);
    atomicAdd(&agg[(size_t)d * HC + t], feat[(size_t)s * HC + t] * alpha);
}

__global__ __launch_bounds__(256) void k_bn_stats(
        const float* __restrict__ agg, const float* __restrict__ gamma,
        const float* __restrict__ beta, float* __restrict__ scale,
        float* __restrict__ shift)
{
    __shared__ float ss[256], sq[256];
    int c = blockIdx.x, t = threadIdx.x;
    float s = 0.0f, q = 0.0f;
    for (int n = t; n < NN; n += 256) {
        float v = agg[(size_t)n * HC + c];
        s += v; q += v * v;
    }
    ss[t] = s; sq[t] = q; __syncthreads();
    for (int o = 128; o; o >>= 1) {
        if (t < o) { ss[t] += ss[t + o]; sq[t] += sq[t + o]; }
        __syncthreads();
    }
    if (t == 0) {
        float mean = ss[0] / (float)NN;
        float var  = sq[0] / (float)NN - mean * mean;
        float r    = rsqrtf(var + 1e-5f);
        float sc   = r * gamma[c];
        scale[c] = sc;
        shift[c] = beta[c] - mean * sc;
    }
}

// h1 (bf16) = elu(agg * scale + shift)
__global__ void k_bn_elu(const float* __restrict__ agg, const float* __restrict__ scale,
                         const float* __restrict__ shift, unsigned short* __restrict__ hb)
{
    int i = blockIdx.x * blockDim.x + threadIdx.x;
    if (i >= NN * HC) return;
    int c = i & (HC - 1);
    float y = fmaf(agg[i], scale[c], shift[c]);
    hb[i] = f2bf(y > 0.0f ? y : (__expf(y) - 1.0f));
}

// jk (bf16) = max(h1, h2)
__global__ void k_jk(const unsigned short* __restrict__ hb,
                     const float* __restrict__ h2, unsigned short* __restrict__ jkb)
{
    int i = blockIdx.x * blockDim.x + threadIdx.x;
    if (i >= NN * HC) return;
    jkb[i] = f2bf(fmaxf(bf2f(hb[i]), h2[i]));
}

// ---------------------------------------------------------------------------
extern "C" void kernel_launch(void* const* d_in, const int* in_sizes, int n_in,
                              void* d_out, int out_size, void* d_ws, size_t ws_size,
                              hipStream_t stream) {
    const float*     x     = (const float*)d_in[0];
    const long long* ei    = (const long long*)d_in[1];
    const float*     W1    = (const float*)d_in[2];
    const float*     a1s   = (const float*)d_in[3];
    const float*     a1d   = (const float*)d_in[4];
    const float*     b1    = (const float*)d_in[5];
    const float*     gamma = (const float*)d_in[6];
    const float*     beta  = (const float*)d_in[7];
    const float*     W2    = (const float*)d_in[8];
    const float*     a2s   = (const float*)d_in[9];
    const float*     a2d   = (const float*)d_in[10];
    const float*     b2    = (const float*)d_in[11];
    const float*     linW  = (const float*)d_in[12];
    const float*     linb  = (const float*)d_in[13];
    float*           out   = (float*)d_out;

    // workspace carve-up (256B aligned)
    char* ws = (char*)d_ws;
    unsigned*       Bp1   = (unsigned*)(ws + 0);                // 64 KB
    unsigned*       Bp2   = (unsigned*)(ws + 65536);            // 128 KB
    unsigned*       BpL   = (unsigned*)(ws + 196608);           // 32 KB
    float*          feat  = (float*)(ws + 229376);              // N*256 f32
    float*          agg   = (float*)(ws + 41189376);            // N*256 f32
    float*          esrc  = (float*)(ws + 82149376);            // N*4
    float*          edstv = (float*)(ws + 82789376);            // N*4
    float*          ee    = (float*)(ws + 83429376);            // E*4
    unsigned*       umax  = (unsigned*)(ws + 88549376);         // N*4
    float*          denom = (float*)(ws + 89189376);            // N*4
    float*          scale = (float*)(ws + 89829376);            // 256
    float*          shift = (float*)(ws + 89830400);            // 256
    unsigned short* xb    = (unsigned short*)(ws + 89831424);   // N*128 bf16
    unsigned short* hb    = (unsigned short*)(ws + 100071424);  // N*256 bf16
    unsigned short* jkb   = (unsigned short*)(ws + 120551424);  // N*256 bf16

    const dim3 blk256(256);
    const int gM  = (NN + 127) / 128;
    const int gEH = (EE * HEADS + 255) / 256;
    const int gNH = (NH + 255) / 256;
    const int gNC = (NN * HC + 255) / 256;

    // pack weights into WMMA fragment layout; convert x to bf16
    k_pack_b<<<(FIN * HC / 2 + 255) / 256, blk256, 0, stream>>>(W1, Bp1, FIN, HC);
    k_pack_b<<<(HC * HC / 2 + 255) / 256, blk256, 0, stream>>>(W2, Bp2, HC, HC);
    k_pack_b<<<(HC * OUTC / 2 + 255) / 256, blk256, 0, stream>>>(linW, BpL, HC, OUTC);
    k_cvt_bf16<<<(NN * FIN + 255) / 256, blk256, 0, stream>>>(x, xb, NN * FIN);

    // ---------------- layer 1 ----------------
    k_gemm<FIN, HC><<<dim3(gM, HC / 64), blk256, 0, stream>>>(xb, Bp1, nullptr, feat, NN);
    k_attn<<<NN, 128, 0, stream>>>(feat, a1s, a1d, esrc, edstv);
    k_init_attn<<<gNH, blk256, 0, stream>>>(umax, denom);
    k_edge1<<<gEH, blk256, 0, stream>>>(esrc, edstv, ei, ee, umax);
    k_edge2<<<gEH, blk256, 0, stream>>>(ei, ee, umax, denom);
    k_agg_init<<<gNC, blk256, 0, stream>>>(agg, b1);
    k_scatter<<<EE, blk256, 0, stream>>>(feat, ee, denom, ei, agg);
    k_bn_stats<<<HC, blk256, 0, stream>>>(agg, gamma, beta, scale, shift);
    k_bn_elu<<<gNC, blk256, 0, stream>>>(agg, scale, shift, hb);

    // ---------------- layer 2 ----------------
    k_gemm<HC, HC><<<dim3(gM, HC / 64), blk256, 0, stream>>>(hb, Bp2, nullptr, feat, NN);
    k_attn<<<NN, 128, 0, stream>>>(feat, a2s, a2d, esrc, edstv);
    k_init_attn<<<gNH, blk256, 0, stream>>>(umax, denom);
    k_edge1<<<gEH, blk256, 0, stream>>>(esrc, edstv, ei, ee, umax);
    k_edge2<<<gEH, blk256, 0, stream>>>(ei, ee, umax, denom);
    k_agg_init<<<gNC, blk256, 0, stream>>>(agg, b2);          // agg becomes h2
    k_scatter<<<EE, blk256, 0, stream>>>(feat, ee, denom, ei, agg);

    // ---------------- JK max + final linear ----------------
    k_jk<<<gNC, blk256, 0, stream>>>(hb, agg, jkb);
    k_gemm<HC, OUTC><<<dim3(gM, 1), blk256, 0, stream>>>(jkb, BpL, linb, out, NN);
}
